// MoE_dist_66778151518301
// MI455X (gfx1250) — compile-verified, matrix operand order checked by reading, not run on hardware
//
#include <hip/hip_runtime.h>

// ---------------------------------------------------------------------------
// Problem constants (match reference)
// ---------------------------------------------------------------------------
#define B_    8
#define T_    2048
#define C_    1024
#define E_    8
#define KTOP  2
#define S_    (T_ * KTOP)      // 4096 assignments per batch row
#define CAP_  320
#define DFF_  4096

#define MT    32               // token rows per FFN block
#define HP    (DFF_ + 8)       // padded LDS row stride (bf16 elems)

typedef __attribute__((ext_vector_type(16))) __bf16 bf16x16;
typedef __attribute__((ext_vector_type(8)))  float  f32x8;

union Frag {
    bf16x16 v;
    uint4   u[2];
};

__device__ __forceinline__ unsigned short f2bf(float f) {
    unsigned int u = __float_as_uint(f);
    u += 0x7FFFu + ((u >> 16) & 1u);          // round-to-nearest-even
    return (unsigned short)(u >> 16);
}

// ---------------------------------------------------------------------------
// K1: fp32 -> bf16 convert + transpose: src[z][r][c] -> dst[z][c][r]
// ---------------------------------------------------------------------------
__global__ void k_transpose_bf16(const float* __restrict__ src,
                                 unsigned short* __restrict__ dst,
                                 int R, int Cc) {
    __shared__ float tile[32][33];
    const int z  = blockIdx.z;
    const int c0 = blockIdx.x * 32;
    const int r0 = blockIdx.y * 32;
    const int tx = threadIdx.x, ty = threadIdx.y;
    const float* s = src + (size_t)z * R * Cc;
    unsigned short* d = dst + (size_t)z * R * Cc;
#pragma unroll
    for (int i = 0; i < 4; ++i)
        tile[ty + i * 8][tx] = s[(size_t)(r0 + ty + i * 8) * Cc + (c0 + tx)];
    __syncthreads();
#pragma unroll
    for (int i = 0; i < 4; ++i)
        d[(size_t)(c0 + ty + i * 8) * R + (r0 + tx)] = f2bf(tile[tx][ty + i * 8]);
}

// ---------------------------------------------------------------------------
// K2: router — one wave32 per token: logits, softmax, top-2
// ---------------------------------------------------------------------------
__global__ __launch_bounds__(256)
void k_router(const float* __restrict__ x, const float* __restrict__ Wg,
              const float* __restrict__ bg,
              int* __restrict__ tke, float* __restrict__ tkp) {
    const int wv    = threadIdx.x >> 5;
    const int lane  = threadIdx.x & 31;
    const int token = blockIdx.x * 8 + wv;           // flat b*T + t

    float acc[E_];
#pragma unroll
    for (int e = 0; e < E_; ++e) acc[e] = 0.f;

    const float* xr = x + (size_t)token * C_;
    for (int c = lane; c < C_; c += 32) {
        const float xv = xr[c];
        const float4* wr = (const float4*)(Wg + (size_t)c * E_);
        const float4 w0 = wr[0], w1 = wr[1];
        acc[0] += xv * w0.x; acc[1] += xv * w0.y;
        acc[2] += xv * w0.z; acc[3] += xv * w0.w;
        acc[4] += xv * w1.x; acc[5] += xv * w1.y;
        acc[6] += xv * w1.z; acc[7] += xv * w1.w;
    }
#pragma unroll
    for (int e = 0; e < E_; ++e)
#pragma unroll
        for (int off = 16; off > 0; off >>= 1)
            acc[e] += __shfl_xor(acc[e], off, 32);

    if (lane == 0) {
        float lg[E_];
#pragma unroll
        for (int e = 0; e < E_; ++e) lg[e] = acc[e] + bg[e];
        int e0 = 0;
#pragma unroll
        for (int e = 1; e < E_; ++e) if (lg[e] > lg[e0]) e0 = e;  // ties -> lowest idx
        int e1 = (e0 == 0) ? 1 : 0;
#pragma unroll
        for (int e = 0; e < E_; ++e)
            if (e != e0 && lg[e] > lg[e1]) e1 = e;
        const float mx = lg[e0];
        float denom = 0.f;
#pragma unroll
        for (int e = 0; e < E_; ++e) denom += __expf(lg[e] - mx);
        const float inv = 1.0f / denom;
        tke[token * 2 + 0] = e0;
        tke[token * 2 + 1] = e1;
        tkp[token * 2 + 0] = inv;                        // exp(0)/denom
        tkp[token * 2 + 1] = __expf(lg[e1] - mx) * inv;
    }
}

// ---------------------------------------------------------------------------
// K3: per-(b,e) in-order cumulative positions + capacity drop (pos=-1)
// ---------------------------------------------------------------------------
__global__ void k_positions(const int* __restrict__ tke, int* __restrict__ pos_arr) {
    const int tid = threadIdx.x;
    if (tid >= B_ * E_) return;
    const int b = tid / E_, e = tid % E_;
    const int* eb = tke + (size_t)b * S_;
    int cnt = 0;
    for (int s = 0; s < S_; ++s) {
        if (eb[s] == e) {
            pos_arr[(size_t)b * S_ + s] = (cnt < CAP_) ? cnt : -1;
            ++cnt;
        }
    }
}

// ---------------------------------------------------------------------------
// K4: scatter kept tokens into dispatch buffer [E][B][CAP][C] (bf16)
// ---------------------------------------------------------------------------
__global__ __launch_bounds__(256)
void k_scatter(const float* __restrict__ x, const int* __restrict__ tke,
               const int* __restrict__ pos_arr, unsigned short* __restrict__ disp) {
    const int blk = blockIdx.x;                 // b*S + s
    const int b = blk / S_, s = blk % S_;
    const int pos = pos_arr[(size_t)b * S_ + s];
    if (pos < 0) return;
    const int e = tke[(size_t)b * S_ + s];
    const int t = s >> 1;
    const int c = threadIdx.x * 4;
    const float4 v = *(const float4*)(x + ((size_t)(b * T_ + t)) * C_ + c);
    unsigned short* d = disp + (((size_t)(e * B_ + b) * CAP_ + pos)) * C_ + c;
    ushort4 o;
    o.x = f2bf(v.x); o.y = f2bf(v.y); o.z = f2bf(v.z); o.w = f2bf(v.w);
    *(ushort4*)d = o;
}

// ---------------------------------------------------------------------------
// K5: fused expert FFN:  y = relu(disp @ W1 + b1) @ W2 + b2   (bf16 WMMA)
// One block = (expert e, row b, 32-token tile), 8 waves.
// Full h tile (32 x 4096 bf16 = 256.5 KB) lives in LDS (MI455X: 320 KB/WG).
// Loops are deliberately NOT unrolled beyond the 8-WMMA steady-state body so
// the live set stays ~150 VGPR (no spills, no VGPR-MSB switching).
// ---------------------------------------------------------------------------
__global__ __launch_bounds__(256) __attribute__((amdgpu_waves_per_eu(4)))
void k_expert_ffn(const unsigned short* __restrict__ disp,  // [E][B][CAP][C] bf16
                  const unsigned short* __restrict__ W1t,   // [E][DFF][C]   bf16 (K-major)
                  const unsigned short* __restrict__ W2t,   // [E][C][DFF]   bf16 (K-major)
                  const float* __restrict__ b1,             // [E][DFF]
                  const float* __restrict__ b2,             // [E][C]
                  float* __restrict__ y) {                   // [E][B][CAP][C]
    extern __shared__ unsigned short hsm[];                  // [MT][HP] bf16

    const int blk = blockIdx.x;
    const int e   = blk / (B_ * (CAP_ / MT));
    const int rb  = blk % (B_ * (CAP_ / MT));
    const int b   = rb / (CAP_ / MT);
    const int m0  = (rb % (CAP_ / MT)) * MT;

    const int w    = threadIdx.x >> 5;
    const int lane = threadIdx.x & 31;
    const int col  = lane & 15;          // M row (A) or N col (B/C/D) index
    const int k0   = (lane >> 4) * 8;    // K-half select per §7.12.2 layouts

    const unsigned short* Ae  = disp + (size_t)(e * B_ + b) * CAP_ * C_;
    const unsigned short* W1e = W1t + (size_t)e * DFF_ * C_;
    const unsigned short* W2e = W2t + (size_t)e * C_ * DFF_;

    // A-row base pointers (fixed for whole kernel)
    const unsigned short* a0 = Ae + (size_t)(m0 + col) * C_ + k0;
    const unsigned short* a1 = a0 + (size_t)16 * C_;

    // ---------------- phase A: h = relu(A @ W1 + b1) -> LDS ----------------
    // 256 col-tiles of 16; wave owns 32; 4 tiles per group.
#pragma unroll 1
    for (int g = 0; g < 8; ++g) {
        f32x8 hacc[2][4];
#pragma unroll
        for (int m = 0; m < 2; ++m)
#pragma unroll
            for (int q = 0; q < 4; ++q)
#pragma unroll
                for (int v = 0; v < 8; ++v) hacc[m][q][v] = 0.f;

        const int nt0 = w * 32 + g * 4;                 // first dff col-tile
        const unsigned short* wq[4];
#pragma unroll
        for (int q = 0; q < 4; ++q)
            wq[q] = W1e + (size_t)((nt0 + q) * 16 + col) * C_ + k0;

#pragma unroll 1
        for (int kk = 0; kk < C_; kk += 32) {
            Frag afr0, afr1;
            afr0.u[0] = *(const uint4*)(a0 + kk);
            afr0.u[1] = *(const uint4*)(a0 + kk + 16);
            afr1.u[0] = *(const uint4*)(a1 + kk);
            afr1.u[1] = *(const uint4*)(a1 + kk + 16);
#pragma unroll
            for (int q = 0; q < 4; ++q) {
                Frag bfr;
                bfr.u[0] = *(const uint4*)(wq[q] + kk);
                bfr.u[1] = *(const uint4*)(wq[q] + kk + 16);
                hacc[0][q] = __builtin_amdgcn_wmma_f32_16x16x32_bf16(
                    false, afr0.v, false, bfr.v, (short)0, hacc[0][q], false, false);
                hacc[1][q] = __builtin_amdgcn_wmma_f32_16x16x32_bf16(
                    false, afr1.v, false, bfr.v, (short)0, hacc[1][q], false, false);
            }
        }
#pragma unroll
        for (int q = 0; q < 4; ++q) {
            const int dcol = (nt0 + q) * 16 + col;
            const float bv1 = b1[e * DFF_ + dcol];
#pragma unroll
            for (int m = 0; m < 2; ++m)
#pragma unroll
                for (int v = 0; v < 8; ++v)
                    hsm[(size_t)(m * 16 + k0 + v) * HP + dcol] =
                        f2bf(fmaxf(hacc[m][q][v] + bv1, 0.f));
        }
    }
    __syncthreads();

    // ---------------- phase B: y = h @ W2 + b2 ----------------
    // 64 out col-tiles of 16; wave owns 8; 4 tiles per group.
    const unsigned short* h0 = hsm + (size_t)col * HP + k0;
    const unsigned short* h1 = h0 + (size_t)16 * HP;

#pragma unroll 1
    for (int g = 0; g < 2; ++g) {
        f32x8 acc[2][4];
#pragma unroll
        for (int m = 0; m < 2; ++m)
#pragma unroll
            for (int q = 0; q < 4; ++q)
#pragma unroll
                for (int v = 0; v < 8; ++v) acc[m][q][v] = 0.f;

        const int ct0 = w * 8 + g * 4;                  // first out col-tile
        const unsigned short* wq[4];
#pragma unroll
        for (int q = 0; q < 4; ++q)
            wq[q] = W2e + (size_t)((ct0 + q) * 16 + col) * DFF_ + k0;

#pragma unroll 1
        for (int kk = 0; kk < DFF_; kk += 32) {
            Frag afr0, afr1;
            afr0.u[0] = *(const uint4*)(h0 + kk);
            afr0.u[1] = *(const uint4*)(h0 + kk + 16);
            afr1.u[0] = *(const uint4*)(h1 + kk);
            afr1.u[1] = *(const uint4*)(h1 + kk + 16);
#pragma unroll
            for (int q = 0; q < 4; ++q) {
                Frag bfr;
                bfr.u[0] = *(const uint4*)(wq[q] + kk);
                bfr.u[1] = *(const uint4*)(wq[q] + kk + 16);
                acc[0][q] = __builtin_amdgcn_wmma_f32_16x16x32_bf16(
                    false, afr0.v, false, bfr.v, (short)0, acc[0][q], false, false);
                acc[1][q] = __builtin_amdgcn_wmma_f32_16x16x32_bf16(
                    false, afr1.v, false, bfr.v, (short)0, acc[1][q], false, false);
            }
        }
        // epilogue for this group: + b2, store y (fp32)
#pragma unroll
        for (int q = 0; q < 4; ++q) {
            const int ccol = (ct0 + q) * 16 + col;
            const float bv2 = b2[e * C_ + ccol];
#pragma unroll
            for (int m = 0; m < 2; ++m) {
                float* yp = y + ((size_t)(e * B_ + b) * CAP_ + m0 + m * 16 + k0) * C_ + ccol;
#pragma unroll
                for (int v = 0; v < 8; ++v)
                    yp[(size_t)v * C_] = acc[m][q][v] + bv2;
            }
        }
    }
}

// ---------------------------------------------------------------------------
// K6: combine — out[b,t,:] = sum_k p_k * y[e_k, b, pos_k, :]
// ---------------------------------------------------------------------------
__global__ __launch_bounds__(256)
void k_combine(const float* __restrict__ y, const int* __restrict__ tke,
               const float* __restrict__ tkp, const int* __restrict__ pos_arr,
               float* __restrict__ out) {
    const int blk = blockIdx.x;               // b*T + t
    const int b = blk / T_, t = blk % T_;
    const int c = threadIdx.x * 4;
    float4 o = make_float4(0.f, 0.f, 0.f, 0.f);
#pragma unroll
    for (int k = 0; k < KTOP; ++k) {
        const size_t idx = (size_t)b * S_ + (t * KTOP + k);
        const int pos = pos_arr[idx];
        if (pos >= 0) {
            const int e = tke[idx];
            const float p = tkp[idx];
            const float4 v = *(const float4*)(
                y + (((size_t)(e * B_ + b) * CAP_ + pos)) * C_ + c);
            o.x += p * v.x; o.y += p * v.y; o.z += p * v.z; o.w += p * v.w;
        }
    }
    *(float4*)(out + (size_t)(b * T_ + t) * C_ + c) = o;
}

// ---------------------------------------------------------------------------
// Workspace layout
// ---------------------------------------------------------------------------
static constexpr size_t SZ_W1T  = (size_t)E_ * DFF_ * C_ * 2;     //  64 MiB
static constexpr size_t SZ_W2T  = (size_t)E_ * C_ * DFF_ * 2;     //  64 MiB
static constexpr size_t SZ_DISP = (size_t)E_ * B_ * CAP_ * C_ * 2;//  40 MiB
static constexpr size_t SZ_Y    = (size_t)E_ * B_ * CAP_ * C_ * 4;//  80 MiB
static constexpr size_t SZ_TKE  = (size_t)B_ * S_ * 4;
static constexpr size_t SZ_TKP  = (size_t)B_ * S_ * 4;

static constexpr size_t OFF_W1T  = 0;
static constexpr size_t OFF_W2T  = OFF_W1T + SZ_W1T;
static constexpr size_t OFF_DISP = OFF_W2T + SZ_W2T;
static constexpr size_t OFF_Y    = OFF_DISP + SZ_DISP;
static constexpr size_t OFF_TKE  = OFF_Y + SZ_Y;
static constexpr size_t OFF_TKP  = OFF_TKE + SZ_TKE;
static constexpr size_t OFF_POS  = OFF_TKP + SZ_TKP;

static constexpr size_t LDS_H_BYTES = (size_t)MT * HP * 2;        // 256.5 KiB

extern "C" void kernel_launch(void* const* d_in, const int* in_sizes, int n_in,
                              void* d_out, int out_size, void* d_ws, size_t ws_size,
                              hipStream_t stream) {
    const float* x  = (const float*)d_in[0];
    const float* Wg = (const float*)d_in[1];
    const float* bg = (const float*)d_in[2];
    const float* W1 = (const float*)d_in[3];
    const float* b1 = (const float*)d_in[4];
    const float* W2 = (const float*)d_in[5];
    const float* b2 = (const float*)d_in[6];
    float* out = (float*)d_out;

    unsigned char* ws = (unsigned char*)d_ws;
    unsigned short* W1t  = (unsigned short*)(ws + OFF_W1T);
    unsigned short* W2t  = (unsigned short*)(ws + OFF_W2T);
    unsigned short* disp = (unsigned short*)(ws + OFF_DISP);
    float*          y    = (float*)(ws + OFF_Y);
    int*            tke  = (int*)(ws + OFF_TKE);
    float*          tkp  = (float*)(ws + OFF_TKP);
    int*            pos  = (int*)(ws + OFF_POS);

    // 1) weights: fp32 -> bf16, K-major transpose
    k_transpose_bf16<<<dim3(DFF_ / 32, C_ / 32, E_), dim3(32, 8), 0, stream>>>(
        W1, W1t, C_, DFF_);
    k_transpose_bf16<<<dim3(C_ / 32, DFF_ / 32, E_), dim3(32, 8), 0, stream>>>(
        W2, W2t, DFF_, C_);

    // 2) router + top-2
    k_router<<<(B_ * T_) / 8, 256, 0, stream>>>(x, Wg, bg, tke, tkp);

    // 3) in-order positions with capacity drop
    k_positions<<<1, 64, 0, stream>>>(tke, pos);

    // 4) dispatch (zero then scatter; dropped slots stay zero)
    hipMemsetAsync(disp, 0, SZ_DISP, stream);
    k_scatter<<<B_ * S_, 256, 0, stream>>>(x, tke, pos, disp);

    // 5) fused expert FFN (bf16 WMMA, h tile resident in 256.5 KB LDS)
    k_expert_ffn<<<E_ * B_ * (CAP_ / MT), 256, LDS_H_BYTES, stream>>>(
        disp, W1t, W2t, b1, b2, y);

    // 6) combine
    k_combine<<<B_ * T_, 256, 0, stream>>>(y, tke, tkp, pos, out);
}
